// MoEMLP_790273982481
// MI455X (gfx1250) — compile-verified, matrix operand order checked by reading, not run on hardware
//
#include <hip/hip_runtime.h>
#include <hip/hip_bf16.h>

typedef __attribute__((ext_vector_type(16))) __bf16 v16bf;
typedef __attribute__((ext_vector_type(8)))  float  v8f;

union BfVec {
    v16bf v;
    uint4 q[2];
};

__device__ __forceinline__ unsigned short f2bf(float f) {
    unsigned u = __float_as_uint(f);
    unsigned r = u + 0x7fffu + ((u >> 16) & 1u);   // round-to-nearest-even
    return (unsigned short)(r >> 16);
}

__device__ __forceinline__ float gelu_exact(float v) {
    return 0.5f * v * (1.0f + erff(v * 0.7071067811865476f));
}

#define AP 264   // A tile row stride in ushorts (256 + 8 pad -> conflict-free LDS)
#define HP 520   // H tile row stride in ushorts (512 + 8 pad)

// ---------------------------------------------------------------------------
// Stage 2: convert weights to bf16 in the *native WMMA B-fragment layout*.
// For each 16x16 output tile (c) and k-step (kk), lane l's 16 bf16 fragment
// elements are stored contiguously (32 B), so a wave's B load is one fully
// coalesced 1 KB burst.
//   W1t element offset: (((e*32+c)*8  + kk)*32 + lane)*16 + i   (K=256, H cols)
//   W2t element offset: (((e*16+c)*16 + kk)*32 + lane)*16 + i   (K=512, DOUT cols)
// Fragment: lm = lane&15, kb = kk*32 + (lane>=16 ? 8 : 0),
//           k  = kb + i + (i<8 ? 0 : 8)   (chunks k+{0..7}, k+{16..23})
// ---------------------------------------------------------------------------
__global__ void moe_convert_kernel(const float* __restrict__ W1,
                                   const float* __restrict__ W2,
                                   unsigned short* __restrict__ W1t,
                                   unsigned short* __restrict__ W2t) {
    const int n1 = 1 << 20;   // 8*256*512
    int idx = blockIdx.x * 256 + threadIdx.x;
    if (idx < n1) {
        int i    = idx & 15;
        int lane = (idx >> 4) & 31;
        int kk   = (idx >> 9) & 7;
        int c    = (idx >> 12) & 31;
        int e    = idx >> 17;
        int lm   = lane & 15;
        int kb   = kk * 32 + (lane >= 16 ? 8 : 0);
        int k    = kb + i + (i < 8 ? 0 : 8);
        int h    = c * 16 + lm;
        W1t[idx] = f2bf(W1[(e * 256 + k) * 512 + h]);
    } else {
        int j    = idx - n1;
        int i    = j & 15;
        int lane = (j >> 4) & 31;
        int kk   = (j >> 9) & 15;
        int c    = (j >> 13) & 15;
        int e    = j >> 17;
        int lm   = lane & 15;
        int kb   = kk * 32 + (lane >= 16 ? 8 : 0);
        int k    = kb + i + (i < 8 ? 0 : 8);
        int n    = c * 16 + lm;
        W2t[j] = f2bf(W2[(e * 512 + k) * 256 + n]);
    }
}

// ---------------------------------------------------------------------------
// Stage 3: LayerNorm + router (top-2 softmax gate), one wave32 per token.
// ---------------------------------------------------------------------------
__global__ void __launch_bounds__(256)
moe_ln_router_kernel(const float* __restrict__ x,
                     const float* __restrict__ gamma,
                     const float* __restrict__ beta,
                     const float* __restrict__ Wr,
                     const float* __restrict__ br,
                     unsigned short* __restrict__ xn,
                     int* __restrict__ counts,
                     int* __restrict__ btok,
                     float* __restrict__ bw,
                     int N) {
    int wave = threadIdx.x >> 5;
    int lane = threadIdx.x & 31;
    int tok = blockIdx.x * 8 + wave;
    if (tok >= N) return;

    const float* xr = x + (size_t)tok * 256 + lane * 8;
    float4 a0 = ((const float4*)xr)[0];
    float4 a1 = ((const float4*)xr)[1];
    float v[8] = {a0.x, a0.y, a0.z, a0.w, a1.x, a1.y, a1.z, a1.w};

    float s = 0.f, sq = 0.f;
#pragma unroll
    for (int i = 0; i < 8; i++) { s += v[i]; sq += v[i] * v[i]; }
#pragma unroll
    for (int off = 16; off >= 1; off >>= 1) {
        s  += __shfl_xor(s,  off, 32);
        sq += __shfl_xor(sq, off, 32);
    }
    float mean = s * (1.f / 256.f);
    float var  = sq * (1.f / 256.f) - mean * mean;
    float rstd = rsqrtf(var + 1e-5f);

    const float4* g4 = (const float4*)(gamma + lane * 8);
    const float4* b4 = (const float4*)(beta  + lane * 8);
    float4 g0 = g4[0], g1 = g4[1], e0 = b4[0], e1 = b4[1];
    float gg[8] = {g0.x, g0.y, g0.z, g0.w, g1.x, g1.y, g1.z, g1.w};
    float bb[8] = {e0.x, e0.y, e0.z, e0.w, e1.x, e1.y, e1.z, e1.w};

    float acc[8];
#pragma unroll
    for (int e = 0; e < 8; e++) acc[e] = 0.f;

    union { unsigned short us[8]; uint4 q; } obf;
#pragma unroll
    for (int i = 0; i < 8; i++) {
        float xni = (v[i] - mean) * rstd * gg[i] + bb[i];
        obf.us[i] = f2bf(xni);
        const float* wrow = Wr + (lane * 8 + i) * 8;
#pragma unroll
        for (int e = 0; e < 8; e++) acc[e] += xni * wrow[e];
    }
    ((uint4*)(xn + (size_t)tok * 256 + lane * 8))[0] = obf.q;

#pragma unroll
    for (int e = 0; e < 8; e++) {
#pragma unroll
        for (int off = 16; off >= 1; off >>= 1)
            acc[e] += __shfl_xor(acc[e], off, 32);
    }

    if (lane == 0) {
        float best1 = -3.4e38f, best2 = -3.4e38f;
        int i1 = 0, i2 = 0;
#pragma unroll
        for (int e = 0; e < 8; e++) {
            float l = acc[e] + br[e];
            if (l > best1) { best2 = best1; i2 = i1; best1 = l; i1 = e; }
            else if (l > best2) { best2 = l; i2 = e; }
        }
        float p2 = __expf(best2 - best1);
        float denom = 1.f + p2;
        float w1 = 1.f / denom;
        float w2 = p2 / denom;
        int p = atomicAdd(&counts[i1], 1);
        btok[i1 * N + p] = tok;
        bw[i1 * N + p]   = w1;
        p = atomicAdd(&counts[i2], 1);
        btok[i2 * N + p] = tok;
        bw[i2 * N + p]   = w2;
    }
}

// ---------------------------------------------------------------------------
// Stage 4: gathered expert FFN. One block = 32 tokens of one expert bucket.
// 8 wave32s. k-loop outermost: each A fragment pair feeds all column tiles,
// each coalesced B fragment feeds both row tiles (2 WMMAs per B load).
// ---------------------------------------------------------------------------
__global__ void __launch_bounds__(256)
moe_expert_kernel(const unsigned short* __restrict__ xn,
                  const unsigned short* __restrict__ W1t,
                  const unsigned short* __restrict__ W2t,
                  const float* __restrict__ b1,
                  const float* __restrict__ b2,
                  const int* __restrict__ counts,
                  const int* __restrict__ btok,
                  const float* __restrict__ bw,
                  float* __restrict__ y,
                  int N) {
    __shared__ __align__(16) unsigned short A[32 * AP];    // xn tile (padded rows)
    __shared__ __align__(16) unsigned short Hs[32 * HP];   // hidden tile (padded rows)
    __shared__ int   stok[32];
    __shared__ float sw[32];

    const int e = blockIdx.y;
    const int count = counts[e];
    const int row0 = blockIdx.x * 32;
    if (row0 >= count) return;

    const int tid = threadIdx.x;
    if (tid < 32) {
        int slot = row0 + tid;
        if (slot < count) {
            stok[tid] = btok[e * N + slot];
            sw[tid]   = bw[e * N + slot];
        } else {
            stok[tid] = 0;
            sw[tid]   = 0.f;   // padded rows contribute nothing
        }
    }
    __syncthreads();

    // Gather A tile: 32 rows x 256 bf16. 8 threads/row, 4x uint4 each.
    {
        int r = tid >> 3;
        int j = tid & 7;
        const uint4* s4 = (const uint4*)(xn + (size_t)stok[r] * 256);
        uint4* d4 = (uint4*)(A + r * AP);
#pragma unroll
        for (int k = 0; k < 4; k++) d4[j * 4 + k] = s4[j * 4 + k];
    }
    __syncthreads();

    const int wave = tid >> 5;
    const int lane = tid & 31;
    const int lm = lane & 15;
    const bool hi = lane >= 16;

    // ---------------- FC1: [32x256] x [256x512] -> GELU -> Hs ----------------
    {
        v8f acc[4][2];
#pragma unroll
        for (int ci = 0; ci < 4; ci++)
#pragma unroll
            for (int r = 0; r < 2; r++)
                acc[ci][r] = (v8f){0.f, 0.f, 0.f, 0.f, 0.f, 0.f, 0.f, 0.f};

#pragma unroll
        for (int kk = 0; kk < 8; kk++) {
            int kbase = kk * 32 + (hi ? 8 : 0);
            BfVec av0, av1;
            const uint4* a0 = (const uint4*)(A + lm * AP + kbase);
            av0.q[0] = a0[0]; av0.q[1] = a0[2];
            const uint4* a1 = (const uint4*)(A + (16 + lm) * AP + kbase);
            av1.q[0] = a1[0]; av1.q[1] = a1[2];
#pragma unroll
            for (int ci = 0; ci < 4; ci++) {
                int c = wave + ci * 8;
                const uint4* bp = (const uint4*)W1t +
                                  ((size_t)((e * 32 + c) * 8 + kk)) * 64 + lane * 2;
                BfVec bv;
                bv.q[0] = bp[0]; bv.q[1] = bp[1];
                acc[ci][0] = __builtin_amdgcn_wmma_f32_16x16x32_bf16(
                    false, av0.v, false, bv.v, (short)0, acc[ci][0], false, false);
                acc[ci][1] = __builtin_amdgcn_wmma_f32_16x16x32_bf16(
                    false, av1.v, false, bv.v, (short)0, acc[ci][1], false, false);
            }
        }
#pragma unroll
        for (int ci = 0; ci < 4; ci++) {
            int col = (wave + ci * 8) * 16 + lm;
            float bias = b1[e * 512 + col];
#pragma unroll
            for (int r = 0; r < 2; r++) {
#pragma unroll
                for (int j = 0; j < 8; j++) {
                    int m = r * 16 + (hi ? j + 8 : j);
                    Hs[m * HP + col] = f2bf(gelu_exact(acc[ci][r][j] + bias));
                }
            }
        }
    }
    __syncthreads();

    // ---------------- FC2: [32x512] x [512x256], gate scale, atomic add -----
    {
        v8f acc[2][2];
#pragma unroll
        for (int ci = 0; ci < 2; ci++)
#pragma unroll
            for (int r = 0; r < 2; r++)
                acc[ci][r] = (v8f){0.f, 0.f, 0.f, 0.f, 0.f, 0.f, 0.f, 0.f};

#pragma unroll
        for (int kk = 0; kk < 16; kk++) {
            int kbase = kk * 32 + (hi ? 8 : 0);
            BfVec av0, av1;
            const uint4* a0 = (const uint4*)(Hs + lm * HP + kbase);
            av0.q[0] = a0[0]; av0.q[1] = a0[2];
            const uint4* a1 = (const uint4*)(Hs + (16 + lm) * HP + kbase);
            av1.q[0] = a1[0]; av1.q[1] = a1[2];
#pragma unroll
            for (int ci = 0; ci < 2; ci++) {
                int c = wave + ci * 8;
                const uint4* bp = (const uint4*)W2t +
                                  ((size_t)((e * 16 + c) * 16 + kk)) * 64 + lane * 2;
                BfVec bv;
                bv.q[0] = bp[0]; bv.q[1] = bp[1];
                acc[ci][0] = __builtin_amdgcn_wmma_f32_16x16x32_bf16(
                    false, av0.v, false, bv.v, (short)0, acc[ci][0], false, false);
                acc[ci][1] = __builtin_amdgcn_wmma_f32_16x16x32_bf16(
                    false, av1.v, false, bv.v, (short)0, acc[ci][1], false, false);
            }
        }
#pragma unroll
        for (int ci = 0; ci < 2; ci++) {
            int col = (wave + ci * 8) * 16 + lm;
            float bias = b2[e * 256 + col];
#pragma unroll
            for (int r = 0; r < 2; r++) {
#pragma unroll
                for (int j = 0; j < 8; j++) {
                    int rr = r * 16 + (hi ? j + 8 : j);
                    float w = sw[rr];
                    if (w != 0.f) {
                        float val = w * (acc[ci][r][j] + bias);
                        __hip_atomic_fetch_add(&y[(size_t)stok[rr] * 256 + col], val,
                                               __ATOMIC_RELAXED, __HIP_MEMORY_SCOPE_AGENT);
                    }
                }
            }
        }
    }
}

// ---------------------------------------------------------------------------
extern "C" void kernel_launch(void* const* d_in, const int* in_sizes, int n_in,
                              void* d_out, int out_size, void* d_ws, size_t ws_size,
                              hipStream_t stream) {
    const float* x     = (const float*)d_in[0];
    const float* gamma = (const float*)d_in[1];
    const float* beta  = (const float*)d_in[2];
    const float* Wr    = (const float*)d_in[3];
    const float* br    = (const float*)d_in[4];
    const float* W1    = (const float*)d_in[5];
    const float* b1    = (const float*)d_in[6];
    const float* W2    = (const float*)d_in[7];
    const float* b2    = (const float*)d_in[8];
    float* y = (float*)d_out;

    const int N = in_sizes[0] / 256;   // total tokens (B*S)

    char* ws = (char*)d_ws;
    size_t off = 0;
    unsigned short* xn  = (unsigned short*)(ws + off); off += (size_t)N * 256 * 2;
    unsigned short* W1t = (unsigned short*)(ws + off); off += (size_t)8 * 512 * 256 * 2;
    unsigned short* W2t = (unsigned short*)(ws + off); off += (size_t)8 * 256 * 512 * 2;
    int*   counts = (int*)(ws + off);                  off += 256;
    int*   btok   = (int*)(ws + off);                  off += (size_t)8 * N * 4;
    float* bw     = (float*)(ws + off);                off += (size_t)8 * N * 4;

    hipMemsetAsync(counts, 0, 8 * sizeof(int), stream);
    hipMemsetAsync(d_out, 0, (size_t)out_size * sizeof(float), stream);

    moe_convert_kernel<<<(2 * (1 << 20)) / 256, 256, 0, stream>>>(W1, W2, W1t, W2t);

    moe_ln_router_kernel<<<(N + 7) / 8, 256, 0, stream>>>(
        x, gamma, beta, Wr, br, xn, counts, btok, bw, N);

    dim3 grid2((N + 31) / 32, 8);
    moe_expert_kernel<<<grid2, 256, 0, stream>>>(
        xn, W1t, W2t, b1, b2, counts, btok, bw, y, N);
}
